// dcn_v2_84275848282127
// MI455X (gfx1250) — compile-verified, matrix operand order checked by reading, not run on hardware
//
#include <hip/hip_runtime.h>

// ---------------------------------------------------------------------------
// DCNv2 forward as implicit GEMM on CDNA5 WMMA (bf16 in, f32 accumulate).
//   M = B*H*W = 32768 pixels, N = O = 128, K = 9 taps * 128 ch = 1152
//   GEMM K index = t*128 + c. A chunk (K=32) never crosses a tap boundary:
//   t = kc>>2 uniform per chunk, c = (kc&3)*32 + kr -> tap metadata is
//   register-cached for 4 chunks and channel offsets fold into inst offsets.
// ---------------------------------------------------------------------------

typedef __attribute__((ext_vector_type(16))) __bf16 v16bf;
typedef __attribute__((ext_vector_type(2)))  __bf16 v2bf;
typedef __attribute__((ext_vector_type(8)))  float  v8f;

#define BATCH   8
#define CIN     128
#define COUT    128
#define HH      64
#define WW      64
#define HW      4096
#define KK      9
#define KDIM    1152               // 9 * 128
#define KCH     32                 // WMMA K per step (bf16)
#define NKC     36                 // 1152 / 32
#define MTILE   64                 // pixels per block
#define NTILES  8                  // 128 / 16
#define OPAD    68                 // outbuf row stride (bank-conflict pad)

// f32 pair -> packed bf16 dword (RNE): single v_cvt_pk_bf16_f32 (probe-verified)
__device__ __forceinline__ unsigned int pack_bf16(float a, float b) {
#if __has_builtin(__builtin_amdgcn_cvt_pk_bf16_f32)
    v2bf r = __builtin_amdgcn_cvt_pk_bf16_f32(a, b);
    return __builtin_bit_cast(unsigned int, r);
#else
    unsigned int d;
    asm("v_cvt_pk_bf16_f32 %0, %1, %2" : "=v"(d) : "v"(a), "v"(b));
    return d;
#endif
}

// ---------------------------------------------------------------------------
// Prep: weight [O, C, 3, 3] f32 -> bf16 B-fragments in WMMA lane layout,
// with GEMM K index = t*128 + c.
// Bmat dword index = ((kc*8 + ntile)*32 + lane)*8 + j
// ---------------------------------------------------------------------------
__global__ __launch_bounds__(256) void dcn_prep_b(const float* __restrict__ weight,
                                                  unsigned int* __restrict__ Bmat) {
    int idx = blockIdx.x * 256 + threadIdx.x;          // 0 .. 36*8*32*8-1
    if (idx >= NKC * NTILES * 32 * 8) return;
    int j    =  idx        & 7;
    int lane = (idx >> 3)  & 31;
    int nt   = (idx >> 8)  & 7;
    int kc   =  idx >> 11;
    int n    = nt * 16 + (lane & 15);
    int kr   = ((lane & 16) ? 8 : 0) + ((j < 4) ? 2 * j : 16 + 2 * (j - 4));
    int k    = kc * KCH + kr;                          // k even -> (c, c+1), same tap
    int t    = k >> 7;
    int c    = k & 127;
    float v0 = weight[n * KDIM + c * KK + t];
    float v1 = weight[n * KDIM + (c + 1) * KK + t];
    Bmat[idx] = pack_bf16(v0, v1);
}

// ---------------------------------------------------------------------------
// Main fused kernel: one block = 64 pixels x 128 output channels.
// ---------------------------------------------------------------------------
__global__ __launch_bounds__(256) void dcn_main(const float* __restrict__ x,
                                                const float* __restrict__ offsets,
                                                const float* __restrict__ mask,
                                                const unsigned int* __restrict__ Bmat,
                                                const float* __restrict__ bias,
                                                float* __restrict__ out) {
    // GEMM-phase LDS overlaid with epilogue transpose buffer
    union SMem {
        struct {
            uint4        offs[MTILE * KK];   // 4 corner element-offsets (y*64+x)
            float4       wts[MTILE * KK];    // mask-folded, validity-zeroed weights
            unsigned int A[2][4 * 32 * 8];   // bf16 A-fragments, double buffered
        } g;
        float ob[COUT * OPAD];               // [n][p] transpose buffer
    };
    __shared__ __align__(32) SMem sm;

    const int tid  = threadIdx.x;
    const int pg0  = blockIdx.x * MTILE;               // global pixel base
    const int bImg = pg0 >> 12;                        // batch (tile never crosses)

    // ---- phase 0: bilinear metadata -------------------------------------
    for (int e = tid; e < MTILE * KK; e += 256) {
        int p  = e / KK, t = e - p * KK;
        int pg = pg0 + p;
        int hw = pg & (HW - 1);
        int h  = hw >> 6, w = hw & 63;
        float offy = offsets[(bImg * 18 + 2 * t)     * HW + hw];
        float offx = offsets[(bImg * 18 + 2 * t + 1) * HW + hw];
        float mk   = mask   [(bImg *  9 + t)         * HW + hw];
        float py = (float)(h - 1 + t / 3) + offy;
        float px = (float)(w - 1 + t % 3) + offx;
        float y0f = floorf(py), x0f = floorf(px);
        float fy = py - y0f,    fx = px - x0f;
        int y0 = (int)y0f, x0 = (int)x0f;
        int y1 = y0 + 1,   x1 = x0 + 1;
        float vy0 = (y0 >= 0 && y0 < HH) ? 1.f : 0.f;
        float vy1 = (y1 >= 0 && y1 < HH) ? 1.f : 0.f;
        float vx0 = (x0 >= 0 && x0 < WW) ? 1.f : 0.f;
        float vx1 = (x1 >= 0 && x1 < WW) ? 1.f : 0.f;
        float4 wv;
        wv.x = (1.f - fy) * (1.f - fx) * mk * vy0 * vx0;
        wv.y = (1.f - fy) * fx         * mk * vy0 * vx1;
        wv.z = fy         * (1.f - fx) * mk * vy1 * vx0;
        wv.w = fy         * fx         * mk * vy1 * vx1;
        int y0c = min(max(y0, 0), HH - 1);
        int x0c = min(max(x0, 0), WW - 1);
        int y1c = min(max(y1, 0), HH - 1);
        int x1c = min(max(x1, 0), WW - 1);
        uint4 ov;
        ov.x = (unsigned int)(y0c * WW + x0c);
        ov.y = (unsigned int)(y0c * WW + x1c);
        ov.z = (unsigned int)(y1c * WW + x0c);
        ov.w = (unsigned int)(y1c * WW + x1c);
        sm.g.offs[e] = ov;
        sm.g.wts[e]  = wv;
    }
    __syncthreads();

    // ---- per-thread gen invariants (4 dwords/thread/chunk) --------------
    int pI[4], ldsIdx[4], krs[4];
#pragma unroll
    for (int i = 0; i < 4; ++i) {
        int idx = tid + i * 256;                       // 0..1023 dwords
        int p   = idx >> 4;                            // pixel in tile
        int kr  = (idx & 15) * 2;                      // even K row in chunk
        int j   = ((kr & 16) ? 4 : 0) + ((kr & 7) >> 1);
        int ln  = (p & 15) + ((kr & 8) ? 16 : 0);
        pI[i]   = p;
        krs[i]  = kr;
        ldsIdx[i] = (p >> 4) * 256 + ln * 8 + j;
    }

    // register-cached tap metadata: gather voffsets (kr*HW + corner) + weights
    int    vb[4][4];
    float4 mwc[4];
    auto loadMeta = [&](int t) {
#pragma unroll
        for (int i = 0; i < 4; ++i) {
            int mi = pI[i] * KK + t;
            uint4 o = sm.g.offs[mi];
            mwc[i]  = sm.g.wts[mi];
            vb[i][0] = (int)o.x + (krs[i] << 12);
            vb[i][1] = (int)o.y + (krs[i] << 12);
            vb[i][2] = (int)o.z + (krs[i] << 12);
            vb[i][3] = (int)o.w + (krs[i] << 12);
        }
    };

    const float* __restrict__ xb = x + ((size_t)bImg * CIN) * HW;
    // qn = chunk-within-tap (compile-time) -> channel offset = qn*32*HW folds
    // into the load instruction offset; c+1 is a further +HW inst offset.
    auto gen = [&](int qn, int buf) {
#pragma unroll
        for (int i = 0; i < 4; ++i) {
            const int ofs = qn * (32 * HW);
            float4 wv = mwc[i];
            float v0 = wv.x * xb[vb[i][0] + ofs]      + wv.y * xb[vb[i][1] + ofs]
                     + wv.z * xb[vb[i][2] + ofs]      + wv.w * xb[vb[i][3] + ofs];
            float v1 = wv.x * xb[vb[i][0] + ofs + HW] + wv.y * xb[vb[i][1] + ofs + HW]
                     + wv.z * xb[vb[i][2] + ofs + HW] + wv.w * xb[vb[i][3] + ofs + HW];
            sm.g.A[buf][ldsIdx[i]] = pack_bf16(v0, v1);
        }
    };

    // ---- wave tiling: 8 waves = 4 m-tiles x 2 n-halves, 4 acc each ------
    const int wave = tid >> 5, lane = tid & 31;
    const int mtile = wave & 3;
    const int ntb   = (wave >> 2) * 4;
    v8f acc[4] = {v8f{}, v8f{}, v8f{}, v8f{}};

    loadMeta(0);
    gen(0, 0);
    __syncthreads();

    // per-wave B-fragment stream: base pointer + one increment per tap; the
    // (q, a) selection folds into load instruction offsets (max 27 KB).
    const unsigned int* __restrict__ bwp = Bmat + (ntb * 32 + lane) * 8;

    for (int tt = 0; tt < KK; ++tt) {                  // 9 taps
#pragma unroll
        for (int q = 0; q < 4; ++q) {                  // 4 chunks per tap
            const int kc = tt * 4 + q;
            // fragment loads first: latency hides under gen(); loads retire
            // in counter order so WMMA starts once fragments land.
            v16bf af = *(const v16bf*)&sm.g.A[(unsigned)q & 1][mtile * 256 + lane * 8];
            v16bf bfv[4];
#pragma unroll
            for (int a = 0; a < 4; ++a)
                bfv[a] = *(const v16bf*)(bwp + (q * NTILES + a) * 256);

            if (kc + 1 < NKC) {
                if (q == 3) loadMeta(tt + 1);          // next tap's metadata
                gen((q + 1) & 3, (q + 1) & 1);         // pipeline: build chunk kc+1
            }
#pragma unroll
            for (int a = 0; a < 4; ++a)
                acc[a] = __builtin_amdgcn_wmma_f32_16x16x32_bf16(
                    false, af, false, bfv[a], (short)0, acc[a], false, false);
            __syncthreads();
        }
        bwp += 4 * NTILES * 256;                       // advance one tap (32 KB)
    }

    // ---- epilogue: transpose via LDS for coalesced float4 stores --------
#pragma unroll
    for (int a = 0; a < 4; ++a) {
        int n  = (ntb + a) * 16 + (lane & 15);
        int p0 = mtile * 16 + ((lane >= 16) ? 8 : 0);
#pragma unroll
        for (int r = 0; r < 8; ++r)
            sm.ob[n * OPAD + p0 + r] = acc[a][r];
    }
    __syncthreads();

    {
        int n    = tid >> 1;                           // output channel
        int half = tid & 1;                            // 32-pixel half
        float bv = bias[n];
        int hw0  = pg0 & (HW - 1);
        float* op = out + (((size_t)(bImg * COUT + n)) << 12) + hw0 + half * 32;
        const float* lp = &sm.ob[n * OPAD + half * 32];
#pragma unroll
        for (int i = 0; i < 8; ++i) {
            float4 v = *(const float4*)(lp + i * 4);
            v.x += bv; v.y += bv; v.z += bv; v.w += bv;
            *(float4*)(op + i * 4) = v;
        }
    }
}

// ---------------------------------------------------------------------------
extern "C" void kernel_launch(void* const* d_in, const int* in_sizes, int n_in,
                              void* d_out, int out_size, void* d_ws, size_t ws_size,
                              hipStream_t stream) {
    const float* x       = (const float*)d_in[0];
    const float* offsets = (const float*)d_in[1];
    const float* mask    = (const float*)d_in[2];
    const float* weight  = (const float*)d_in[3];
    const float* bias    = (const float*)d_in[4];
    float* out = (float*)d_out;

    unsigned int* Bmat = (unsigned int*)d_ws;          // 36*8*32*8 dwords = 288 KB

    const int bElems = NKC * NTILES * 32 * 8;          // 73728 dwords
    dcn_prep_b<<<(bElems + 255) / 256, 256, 0, stream>>>(weight, Bmat);

    const int mBlocks = (BATCH * HW) / MTILE;          // 512
    dcn_main<<<mBlocks, 256, 0, stream>>>(x, offsets, mask, Bmat, bias, out);
}
